// QuantumMultiHeadSelfAttention_69337952026827
// MI455X (gfx1250) — compile-verified, matrix operand order checked by reading, not run on hardware
//
#include <hip/hip_runtime.h>

// ---------------------------------------------------------------------------
// QuantumMultiHeadSelfAttention on MI455X (gfx1250, wave32, WMMA bf16 + TDM)
// B=2, N=2048, C=1024, H=16, D=64  -> M = B*N = 4096 rows
// ---------------------------------------------------------------------------

typedef __attribute__((ext_vector_type(16))) __bf16       v16bf;
typedef __attribute__((ext_vector_type(8)))  float        v8f;
typedef __attribute__((ext_vector_type(4)))  unsigned int u32x4;
typedef __attribute__((ext_vector_type(8)))  int          i32x8;
typedef __attribute__((ext_vector_type(4)))  int          i32x4;

#define Mrows 4096
#define Cdim  1024
#define Nseq  2048
#define Hn    16
#define Dh    64

__device__ __forceinline__ v8f wmma_bf16(v16bf a, v16bf b, v8f c) {
  return __builtin_amdgcn_wmma_f32_16x16x32_bf16(
      /*neg_a=*/false, a, /*neg_b=*/false, b,
      /*c_mod=*/(short)0, c, /*reuse_a=*/false, /*reuse_b=*/false);
}

// Per-lane async global->LDS staging (ASYNCcnt path).
__device__ __forceinline__ void async_ld_b128(unsigned int lds_addr, const void* gaddr) {
  unsigned long long ga = (unsigned long long)(size_t)gaddr;
  asm volatile("global_load_async_to_lds_b128 %0, %1, off"
               :: "v"(lds_addr), "v"(ga) : "memory");
}

__device__ __forceinline__ void wait_asynccnt0() {
  asm volatile("s_wait_asynccnt 0x0" ::: "memory");
}

// TDM: DMA a 64x64 bf16 tile (contiguous rows, row stride 64 elems) into LDS.
// D# group0: count=1, lds_addr, 57-bit global addr, type=2.
// D# group1: data_size=2B, tensor_dim0=64, tensor_dim1=rows, tile 64x64,
//            tensor_dim0_stride=64. Groups 2/3 zero (2D tensor).
// amdgpu-toolchain (clang-23) builtin form: 6 args, trailing i32x8 + cpol.
__device__ __forceinline__ void tdm_load_tile_64x64_bf16(unsigned int lds_addr,
                                                         const __bf16* gsrc,
                                                         int tensor_rows) {
  unsigned long long ga = (unsigned long long)(size_t)gsrc;
  u32x4 g0;
  g0[0] = 1u;                                            // count=1, user descriptor
  g0[1] = lds_addr;                                      // LDS byte address
  g0[2] = (unsigned int)ga;                              // global addr [31:0]
  g0[3] = (unsigned int)((ga >> 32) & 0x01FFFFFFu) | (2u << 30);  // [56:32] | type=2
  i32x8 g1;
  g1[0] = 0x10000;                                       // data_size=1 (2 bytes)
  g1[1] = (int)(64u << 16);                              // tensor_dim0[15:0]=64
  g1[2] = (int)(((unsigned int)tensor_rows & 0xFFFFu) << 16); // tensor_dim1[15:0]
  g1[3] = (int)(64u << 16);                              // tile_dim0=64
  g1[4] = 64;                                            // tile_dim1=64
  g1[5] = 64;                                            // tensor_dim0_stride=64
  g1[6] = 0;
  g1[7] = 0;
  i32x4 z4 = {0, 0, 0, 0};
  i32x8 z8 = {0, 0, 0, 0, 0, 0, 0, 0};
  __builtin_amdgcn_tensor_load_to_lds(g0, g1, z4, z4, z8, 0);
}

// ---------------------------- conversion kernels ---------------------------

__global__ void cvt_bf16_kernel(const float* __restrict__ s, __bf16* __restrict__ d, int n) {
  int i = blockIdx.x * blockDim.x + threadIdx.x;
  if (i < n) d[i] = (__bf16)s[i];
}

__global__ void copy_f32_kernel(const float* __restrict__ s, float* __restrict__ d, int n) {
  int i = blockIdx.x * blockDim.x + threadIdx.x;
  if (i < n) d[i] = s[i];
}

// Wcomb[o][k] (k in [0,2C)): k<C -> or_w + 0.1*oi_w ; k>=C -> 0.1*or_w - oi_w
__global__ void build_wout_kernel(const float* __restrict__ orw, const float* __restrict__ oiw,
                                  __bf16* __restrict__ d) {
  int i = blockIdx.x * blockDim.x + threadIdx.x;
  if (i < Cdim * 2 * Cdim) {
    int o = i >> 11, k = i & 2047;
    float v;
    if (k < Cdim) v = orw[o * Cdim + k] + 0.1f * oiw[o * Cdim + k];
    else          v = 0.1f * orw[o * Cdim + k - Cdim] - oiw[o * Cdim + k - Cdim];
    d[i] = (__bf16)v;
  }
}

__global__ void build_bias_out_kernel(const float* __restrict__ orb, const float* __restrict__ oib,
                                      float* __restrict__ d) {
  int i = blockIdx.x * blockDim.x + threadIdx.x;
  if (i < Cdim) d[i] = 1.1f * orb[i] - 0.9f * oib[i];
}

// --------------------- projection GEMM (6 fused linears) -------------------
// Y_p = x @ W_p^T + b_p, p = 0..5 (qr,qi,kr,ki,vr,vi), output bf16 in
// head-split layout projbuf[p][b][h][n][d].
// Block tile: 128(M) x 64(O); 8 waves of 32x32 (2x2 WMMA); K step 32.
// Staging via global_load_async_to_lds_b128 (ASYNCcnt).

__global__ __launch_bounds__(256)
void gemm_proj_kernel(const __bf16* __restrict__ xb,
                      const __bf16* __restrict__ wproj,
                      const float*  __restrict__ biasp,
                      __bf16* __restrict__ projbuf) {
  __shared__ __bf16 At[128 * 48];
  __shared__ __bf16 Bt[64 * 48];

  const int tid  = threadIdx.x;
  const int m0   = blockIdx.x * 128;
  const int po0  = blockIdx.y * 64;
  const int p    = po0 >> 10;
  const int o0   = po0 & (Cdim - 1);
  const int wv   = tid >> 5, lane = tid & 31;
  const int half = lane >> 4, lc = lane & 15;
  const int wm0  = (wv >> 1) * 32, wo0 = (wv & 1) * 32;

  v8f acc[2][2];
  for (int i = 0; i < 2; i++)
    for (int j = 0; j < 2; j++) acc[i][j] = v8f{};

  const int lr   = tid >> 2;        // 0..63
  const int lcol = (tid & 3) * 8;   // bf16 column within 32-wide K tile

  const __bf16* Ag = xb + (size_t)m0 * Cdim;
  const __bf16* Bg = wproj + (size_t)p * Cdim * Cdim + (size_t)o0 * Cdim;

  const unsigned int ldsA0 = (unsigned int)(size_t)(void*)&At[lr * 48 + lcol];
  const unsigned int ldsA1 = (unsigned int)(size_t)(void*)&At[(lr + 64) * 48 + lcol];
  const unsigned int ldsB0 = (unsigned int)(size_t)(void*)&Bt[lr * 48 + lcol];

  for (int kt = 0; kt < Cdim; kt += 32) {
    async_ld_b128(ldsA0, &Ag[(size_t)lr * Cdim + kt + lcol]);
    async_ld_b128(ldsA1, &Ag[(size_t)(lr + 64) * Cdim + kt + lcol]);
    async_ld_b128(ldsB0, &Bg[(size_t)lr * Cdim + kt + lcol]);
    if (kt + 32 < Cdim) {
      __builtin_prefetch(&Ag[(size_t)lr * Cdim + kt + 32 + lcol], 0, 1);
      __builtin_prefetch(&Bg[(size_t)lr * Cdim + kt + 32 + lcol], 0, 1);
    }
    wait_asynccnt0();
    __syncthreads();

    v16bf af[2], bfm[2];
    for (int mi = 0; mi < 2; mi++)
      af[mi] = *(const v16bf*)&At[(wm0 + mi * 16 + lc) * 48 + half * 16];
    for (int oi = 0; oi < 2; oi++)
      bfm[oi] = *(const v16bf*)&Bt[(wo0 + oi * 16 + lc) * 48 + half * 16];
    for (int mi = 0; mi < 2; mi++)
      for (int oi = 0; oi < 2; oi++)
        acc[mi][oi] = wmma_bf16(af[mi], bfm[oi], acc[mi][oi]);
    __syncthreads();
  }

  for (int mi = 0; mi < 2; mi++)
    for (int oi = 0; oi < 2; oi++) {
      int o_g = o0 + wo0 + oi * 16 + lc;
      float bb = biasp[(p << 10) + o_g];
      int h = o_g >> 6, d = o_g & 63;
      for (int j = 0; j < 8; j++) {
        int m = m0 + wm0 + mi * 16 + j + 8 * half;
        int bidx = m >> 11, n = m & (Nseq - 1);
        size_t dst = (size_t)p * Mrows * Cdim +
                     ((((size_t)bidx * Hn + h) * Nseq + n) * Dh + d);
        projbuf[dst] = (__bf16)(acc[mi][oi][j] + bb);
      }
    }
}

// ------------------ fused complex-magnitude flash attention ----------------
// Per block: one (b,h), 64 query rows. 4 waves, 16 q-rows each.
// Online softmax over |scores| with 64-key chunks; K tiles staged by the
// Tensor Data Mover (TENSORcnt), V tiles transposed manually into LDS.

__global__ __launch_bounds__(128)
void attn_kernel(const __bf16* __restrict__ projbuf, __bf16* __restrict__ out_ri) {
  __shared__ __bf16 kR[64 * 64], kI[64 * 64];   // [key][d]  (TDM-written, no pad)
  __shared__ __bf16 vRT[64 * 80], vIT[64 * 80]; // [d][key]  (transposed)
  __shared__ __bf16 pb[4][16 * 80];             // per-wave probability tile [qrow][key]

  const int tid = threadIdx.x, wv = tid >> 5, lane = tid & 31;
  const int half = lane >> 4, lc = lane & 15;
  const int bh = blockIdx.x;   // 0..31  (b*16 + h)
  const int qt = blockIdx.y;   // 0..31  (query tile of 64 rows)

  const size_t NB = (size_t)Nseq * Dh;
  const __bf16* qr = projbuf + 0 * (size_t)Mrows * Cdim + (size_t)bh * NB;
  const __bf16* qi = projbuf + 1 * (size_t)Mrows * Cdim + (size_t)bh * NB;
  const __bf16* kr = projbuf + 2 * (size_t)Mrows * Cdim + (size_t)bh * NB;
  const __bf16* ki = projbuf + 3 * (size_t)Mrows * Cdim + (size_t)bh * NB;
  const __bf16* vr = projbuf + 4 * (size_t)Mrows * Cdim + (size_t)bh * NB;
  const __bf16* vi = projbuf + 5 * (size_t)Mrows * Cdim + (size_t)bh * NB;

  const unsigned int ldsKR = (unsigned int)(size_t)(void*)&kR[0];
  const unsigned int ldsKI = (unsigned int)(size_t)(void*)&kI[0];

  // Q fragments straight from global (row length = 128B, 32B-aligned frags)
  const int qrow = qt * 64 + wv * 16 + lc;
  v16bf qrf[2], qif[2], qrn[2];
  for (int s = 0; s < 2; s++) {
    qrf[s] = *(const v16bf*)&qr[(size_t)qrow * Dh + s * 32 + half * 16];
    qif[s] = *(const v16bf*)&qi[(size_t)qrow * Dh + s * 32 + half * 16];
    v16bf t = qrf[s];
    for (int e = 0; e < 16; e++) t[e] = (__bf16)(-(float)t[e]);
    qrn[s] = t;  // -qr for the imaginary score term (bf16 WMMA NEG is C-only)
  }

  float mrun[8], lrun[8];
  v8f oR[4], oI[4];
  for (int j = 0; j < 8; j++) { mrun[j] = -1e30f; lrun[j] = 0.f; }
  for (int t = 0; t < 4; t++) { oR[t] = v8f{}; oI[t] = v8f{}; }

  for (int kc = 0; kc < Nseq / 64; kc++) {
    const int key0 = kc * 64;

    // Stage K chunks via TDM (one wave issues, TENSORcnt-tracked).
    if (wv == 0) {
      tdm_load_tile_64x64_bf16(ldsKR, kr + (size_t)key0 * Dh, Nseq - key0);
      tdm_load_tile_64x64_bf16(ldsKI, ki + (size_t)key0 * Dh, Nseq - key0);
      __builtin_amdgcn_s_wait_tensorcnt(0);
    }
    // Stage V chunks transposed ([d][key]).
    for (int idx = tid; idx < 4096; idx += 128) {
      int r = idx >> 6, d = idx & 63;
      vRT[d * 80 + r] = vr[(size_t)(key0 + r) * Dh + d];
      vIT[d * 80 + r] = vi[(size_t)(key0 + r) * Dh + d];
    }
    __syncthreads();

    // scores: r = qr.kr + qi.ki ; i = qi.kr - qr.ki ; mag = 0.125*sqrt(r^2+i^2)
    float mag[4][8];
    for (int t = 0; t < 4; t++) {
      v8f aR = v8f{}, aI = v8f{};
      for (int s = 0; s < 2; s++) {
        v16bf bR = *(const v16bf*)&kR[(t * 16 + lc) * 64 + s * 32 + half * 16];
        v16bf bI = *(const v16bf*)&kI[(t * 16 + lc) * 64 + s * 32 + half * 16];
        aR = wmma_bf16(qrf[s], bR, aR);
        aR = wmma_bf16(qif[s], bI, aR);
        aI = wmma_bf16(qif[s], bR, aI);
        aI = wmma_bf16(qrn[s], bI, aI);
      }
      for (int j = 0; j < 8; j++) {
        float r = aR[j], im = aI[j];
        mag[t][j] = 0.125f * sqrtf(r * r + im * im);
      }
    }

    // Online softmax update (per-row; rows live in 16-lane halves).
    for (int j = 0; j < 8; j++) {
      float rm = mag[0][j];
      for (int t = 1; t < 4; t++) rm = fmaxf(rm, mag[t][j]);
      for (int msk = 8; msk >= 1; msk >>= 1) rm = fmaxf(rm, __shfl_xor(rm, msk, 32));
      float mn = fmaxf(mrun[j], rm);
      float alpha = __expf(mrun[j] - mn);
      float rs = 0.f;
      for (int t = 0; t < 4; t++) {
        float pe = __expf(mag[t][j] - mn);
        mag[t][j] = pe;
        rs += pe;
      }
      for (int msk = 8; msk >= 1; msk >>= 1) rs += __shfl_xor(rs, msk, 32);
      lrun[j] = alpha * lrun[j] + rs;
      mrun[j] = mn;
      for (int t = 0; t < 4; t++) { oR[t][j] *= alpha; oI[t][j] *= alpha; }
    }

    // C-layout -> A-layout via per-wave LDS round trip.
    for (int t = 0; t < 4; t++)
      for (int j = 0; j < 8; j++)
        pb[wv][(j + 8 * half) * 80 + t * 16 + lc] = (__bf16)mag[t][j];

    v16bf aP[2];
    for (int s = 0; s < 2; s++)
      aP[s] = *(const v16bf*)&pb[wv][lc * 80 + s * 32 + half * 16];

    for (int t = 0; t < 4; t++)
      for (int s = 0; s < 2; s++) {
        v16bf bVr = *(const v16bf*)&vRT[(t * 16 + lc) * 80 + s * 32 + half * 16];
        v16bf bVi = *(const v16bf*)&vIT[(t * 16 + lc) * 80 + s * 32 + half * 16];
        oR[t] = wmma_bf16(aP[s], bVr, oR[t]);
        oI[t] = wmma_bf16(aP[s], bVi, oI[t]);
      }
    __syncthreads();
  }

  // Normalize and emit merged-heads [m][2C] bf16 (real | imag) for final GEMM.
  const int b = bh >> 4, h = bh & 15;
  float invl[8];
  for (int j = 0; j < 8; j++) invl[j] = 1.f / lrun[j];
  for (int t = 0; t < 4; t++)
    for (int j = 0; j < 8; j++) {
      int q = qt * 64 + wv * 16 + j + 8 * half;
      int d = t * 16 + lc;
      size_t m = (size_t)b * Nseq + q;
      out_ri[m * (2 * Cdim) + h * Dh + d]        = (__bf16)(oR[t][j] * invl[j]);
      out_ri[m * (2 * Cdim) + Cdim + h * Dh + d] = (__bf16)(oI[t][j] * invl[j]);
    }
}

// ----------------------- output GEMM (folded complex mix) ------------------
// out = out_ri(4096x2048) @ Wcomb^T(1024x2048) + bias_comb, fp32 to d_out.

__global__ __launch_bounds__(256)
void gemm_out_kernel(const __bf16* __restrict__ ab,
                     const __bf16* __restrict__ wb,
                     const float*  __restrict__ bias,
                     float* __restrict__ out) {
  __shared__ __bf16 At[128 * 48];
  __shared__ __bf16 Bt[64 * 48];

  const int tid = threadIdx.x;
  const int m0 = blockIdx.x * 128, o0 = blockIdx.y * 64;
  const int wv = tid >> 5, lane = tid & 31;
  const int half = lane >> 4, lc = lane & 15;
  const int wm0 = (wv >> 1) * 32, wo0 = (wv & 1) * 32;

  v8f acc[2][2];
  for (int i = 0; i < 2; i++)
    for (int j = 0; j < 2; j++) acc[i][j] = v8f{};

  const int lr = tid >> 2;
  const int lcol = (tid & 3) * 8;
  const int K = 2 * Cdim;

  const unsigned int ldsA0 = (unsigned int)(size_t)(void*)&At[lr * 48 + lcol];
  const unsigned int ldsA1 = (unsigned int)(size_t)(void*)&At[(lr + 64) * 48 + lcol];
  const unsigned int ldsB0 = (unsigned int)(size_t)(void*)&Bt[lr * 48 + lcol];

  for (int kt = 0; kt < K; kt += 32) {
    async_ld_b128(ldsA0, &ab[(size_t)(m0 + lr) * K + kt + lcol]);
    async_ld_b128(ldsA1, &ab[(size_t)(m0 + lr + 64) * K + kt + lcol]);
    async_ld_b128(ldsB0, &wb[(size_t)(o0 + lr) * K + kt + lcol]);
    if (kt + 32 < K) {
      __builtin_prefetch(&ab[(size_t)(m0 + lr) * K + kt + 32 + lcol], 0, 1);
      __builtin_prefetch(&wb[(size_t)(o0 + lr) * K + kt + 32 + lcol], 0, 1);
    }
    wait_asynccnt0();
    __syncthreads();

    v16bf af[2], bfm[2];
    for (int mi = 0; mi < 2; mi++)
      af[mi] = *(const v16bf*)&At[(wm0 + mi * 16 + lc) * 48 + half * 16];
    for (int oi = 0; oi < 2; oi++)
      bfm[oi] = *(const v16bf*)&Bt[(wo0 + oi * 16 + lc) * 48 + half * 16];
    for (int mi = 0; mi < 2; mi++)
      for (int oi = 0; oi < 2; oi++)
        acc[mi][oi] = wmma_bf16(af[mi], bfm[oi], acc[mi][oi]);
    __syncthreads();
  }

  for (int mi = 0; mi < 2; mi++)
    for (int oi = 0; oi < 2; oi++) {
      int o_g = o0 + wo0 + oi * 16 + lc;
      float bb = bias[o_g];
      for (int j = 0; j < 8; j++) {
        int m = m0 + wm0 + mi * 16 + j + 8 * half;
        out[(size_t)m * Cdim + o_g] = acc[mi][oi][j] + bb;
      }
    }
}

// ------------------------------- launcher ----------------------------------

extern "C" void kernel_launch(void* const* d_in, const int* in_sizes, int n_in,
                              void* d_out, int out_size, void* d_ws, size_t ws_size,
                              hipStream_t stream) {
  (void)in_sizes; (void)n_in; (void)out_size; (void)ws_size;

  const float* x = (const float*)d_in[0];
  const float* wq[6]; const float* bq[6];
  for (int p = 0; p < 6; p++) { wq[p] = (const float*)d_in[1 + 2 * p]; bq[p] = (const float*)d_in[2 + 2 * p]; }
  const float* orw = (const float*)d_in[13]; const float* orb = (const float*)d_in[14];
  const float* oiw = (const float*)d_in[15]; const float* oib = (const float*)d_in[16];

  // Workspace layout (all offsets 64B-aligned); total ~92.3 MB.
  char* ws = (char*)d_ws;
  const size_t OFF_XB    = 0;                               // 4096*1024 bf16
  const size_t OFF_WPROJ = OFF_XB    + (size_t)Mrows * Cdim * 2;
  const size_t OFF_WCOMB = OFF_WPROJ + (size_t)6 * Cdim * Cdim * 2;
  const size_t OFF_BPROJ = OFF_WCOMB + (size_t)Cdim * 2 * Cdim * 2;
  const size_t OFF_BCOMB = OFF_BPROJ + (size_t)6 * Cdim * 4;
  const size_t OFF_PROJ  = OFF_BCOMB + (size_t)Cdim * 4 + 192;  // 192B pad, 64B-aligned
  const size_t OFF_OUTRI = OFF_PROJ  + (size_t)6 * Mrows * Cdim * 2;

  __bf16* xb     = (__bf16*)(ws + OFF_XB);
  __bf16* wproj  = (__bf16*)(ws + OFF_WPROJ);
  __bf16* wcomb  = (__bf16*)(ws + OFF_WCOMB);
  float*  biasp  = (float*) (ws + OFF_BPROJ);
  float*  bcomb  = (float*) (ws + OFF_BCOMB);
  __bf16* projbf = (__bf16*)(ws + OFF_PROJ);
  __bf16* outri  = (__bf16*)(ws + OFF_OUTRI);

  // 1) fp32 -> bf16 conversions + folded output weights.
  cvt_bf16_kernel<<<(Mrows * Cdim + 255) / 256, 256, 0, stream>>>(x, xb, Mrows * Cdim);
  for (int p = 0; p < 6; p++) {
    cvt_bf16_kernel<<<(Cdim * Cdim + 255) / 256, 256, 0, stream>>>(wq[p], wproj + (size_t)p * Cdim * Cdim, Cdim * Cdim);
    copy_f32_kernel<<<(Cdim + 255) / 256, 256, 0, stream>>>(bq[p], biasp + p * Cdim, Cdim);
  }
  build_wout_kernel<<<(Cdim * 2 * Cdim + 255) / 256, 256, 0, stream>>>(orw, oiw, wcomb);
  build_bias_out_kernel<<<(Cdim + 255) / 256, 256, 0, stream>>>(orb, oib, bcomb);

  // 2) six fused projections (q/k/v real+imag), head-split bf16 output.
  gemm_proj_kernel<<<dim3(Mrows / 128, 6 * Cdim / 64), 256, 0, stream>>>(xb, wproj, biasp, projbf);

  // 3) fused complex-magnitude flash attention (TDM-staged K tiles).
  attn_kernel<<<dim3(2 * Hn, Nseq / 64), 128, 0, stream>>>(projbf, outri);

  // 4) folded output projection -> fp32 result.
  gemm_out_kernel<<<dim3(Mrows / 128, Cdim / 64), 256, 0, stream>>>(outri, wcomb, bcomb, (float*)d_out);
}
